// GPT_21431886807668
// MI455X (gfx1250) — compile-verified, matrix operand order checked by reading, not run on hardware
//
#include <hip/hip_runtime.h>

// ---------------- problem constants ----------------
#define B_  2
#define T_  1024
#define C_  1024
#define L_  4
#define H_  16
#define HD_ 64
#define V_  50257

typedef __bf16 bf16;
typedef __attribute__((ext_vector_type(16))) __bf16 bf16x16;
typedef __attribute__((ext_vector_type(8)))  __bf16 bf16x8;
typedef __attribute__((ext_vector_type(4)))  __bf16 bf16x4;
typedef __attribute__((ext_vector_type(8)))  float  v8f;

// ---------------- gfx1250 async global->LDS path ----------------
#if defined(__has_builtin)
#  if __has_builtin(__builtin_amdgcn_global_load_async_to_lds_b128)
#    define USE_ASYNC_LDS 1
#  endif
#endif
#ifndef USE_ASYNC_LDS
#  define USE_ASYNC_LDS 0
#endif

// exact parameter type per hipcc diagnostic: int __vector(4) with addrspace quals
typedef int i32x4_vs __attribute__((vector_size(16)));
typedef __attribute__((address_space(1))) i32x4_vs as1_i32x4;
typedef __attribute__((address_space(3))) i32x4_vs as3_i32x4;

// copy 16 bytes (8 bf16) global -> LDS
static __device__ inline void cp16(bf16* lds_dst, const bf16* gsrc) {
#if USE_ASYNC_LDS
    __builtin_amdgcn_global_load_async_to_lds_b128((as1_i32x4*)gsrc,
                                                   (as3_i32x4*)lds_dst, 0, 0);
#else
    *(bf16x8*)lds_dst = *(const bf16x8*)gsrc;
#endif
}

static __device__ inline void async_wait() {
#if USE_ASYNC_LDS
    asm volatile("s_wait_asynccnt 0" ::: "memory");
#endif
}

// ---------------- helpers ----------------
static __device__ inline bf16 f2bf(float f) {
    unsigned u = __builtin_bit_cast(unsigned, f);
    unsigned r = u + 0x7FFFu + ((u >> 16) & 1u);   // round-to-nearest-even
    unsigned short h = (unsigned short)(r >> 16);
    return __builtin_bit_cast(bf16, h);
}

static __device__ inline v8f vzero() {
    v8f z;
    #pragma unroll
    for (int i = 0; i < 8; ++i) z[i] = 0.0f;
    return z;
}

static __device__ inline v8f wmma_bf16(bf16x16 a, bf16x16 b, v8f c) {
    return __builtin_amdgcn_wmma_f32_16x16x32_bf16(false, a, false, b,
                                                   (short)0, c, false, false);
}

// Build a 16-element A/B fragment from two 16-byte chunks
static __device__ inline bf16x16 frag_from(const bf16* p0, const bf16* p1) {
    union { bf16x16 v; bf16x8 h[2]; } u;
    u.h[0] = *(const bf16x8*)p0;
    u.h[1] = *(const bf16x8*)p1;
    return u.v;
}

static __device__ inline float gelu_f(float v) {
    return 0.5f * v * (1.0f + erff(v * 0.70710678118654752f));
}

// ---------------- weight fp32 -> bf16 pre-conversion ----------------
__global__ __launch_bounds__(256) void wconv_kernel(const float* __restrict__ src,
                                                    bf16* __restrict__ dst, int n4) {
    int i = blockIdx.x * 256 + threadIdx.x;
    if (i >= n4) return;
    float4 v = ((const float4*)src)[i];
    bf16x4 o = { f2bf(v.x), f2bf(v.y), f2bf(v.z), f2bf(v.w) };
    ((bf16x4*)dst)[i] = o;
}

// ---------------- embedding: x = wte[idx] + wpe (fp32 residual stream) ------
__global__ __launch_bounds__(256) void embed_kernel(const int* __restrict__ idx,
                                                    const float* __restrict__ wte,
                                                    const float* __restrict__ wpe,
                                                    float* __restrict__ x) {
    int row = blockIdx.x;            // b*T + t
    int t   = row & (T_ - 1);
    int ix  = idx[row];
    int c   = threadIdx.x * 4;
    float4 a = *(const float4*)&wte[(size_t)ix * C_ + c];
    float4 p = *(const float4*)&wpe[(size_t)t  * C_ + c];
    float4 r; r.x = a.x + p.x; r.y = a.y + p.y; r.z = a.z + p.z; r.w = a.w + p.w;
    *(float4*)&x[(size_t)row * C_ + c] = r;
}

// ---------------- layernorm (fp32 in; bf16 or fp32 out) ----------------
template <bool OBF>
__global__ __launch_bounds__(256) void ln_kernel(const float* __restrict__ in,
                                                 void* __restrict__ outv,
                                                 const float* __restrict__ w,
                                                 const float* __restrict__ bias,
                                                 size_t in_row_stride) {
    __shared__ float red[256];
    const int    tid = threadIdx.x;
    const float* xr  = in + (size_t)blockIdx.x * in_row_stride;
    float s = 0.0f;
    for (int c = tid; c < C_; c += 256) s += xr[c];
    red[tid] = s; __syncthreads();
    for (int st = 128; st > 0; st >>= 1) {
        if (tid < st) red[tid] += red[tid + st];
        __syncthreads();
    }
    float mean = red[0] * (1.0f / C_);
    __syncthreads();
    float v = 0.0f;
    for (int c = tid; c < C_; c += 256) { float d = xr[c] - mean; v += d * d; }
    red[tid] = v; __syncthreads();
    for (int st = 128; st > 0; st >>= 1) {
        if (tid < st) red[tid] += red[tid + st];
        __syncthreads();
    }
    float rstd = rsqrtf(red[0] * (1.0f / C_) + 1e-5f);
    for (int c = tid; c < C_; c += 256) {
        float val = (xr[c] - mean) * rstd * w[c] + bias[c];
        if (OBF) ((bf16*)outv)[(size_t)blockIdx.x * C_ + c] = f2bf(val);
        else     ((float*)outv)[(size_t)blockIdx.x * C_ + c] = val;
    }
}

// ---------------- WMMA bf16 GEMM: Y = act(X @ W^T + bias) (+resid) ----------
// X: [M,K] bf16, W: [N,K] bf16 row-major. Output bf16 (OUTBF) or fp32.
// Block: 256 threads = 8 waves (4 in M x 2 in N). Tile 128x128, K-step 32,
// double-buffered LDS fed by async global->LDS copies.
template <int ACT, bool RESID, bool OUTBF>
__global__ __launch_bounds__(256) void gemm_kernel(const bf16* __restrict__ X,
                                                   const bf16* __restrict__ W,
                                                   const float* __restrict__ bias,
                                                   const float* __restrict__ resid,
                                                   void* __restrict__ Yv,
                                                   int M, int N, int K) {
    __shared__ __align__(16) bf16 As[2][128 * 40];  // 128 rows x 32 k (+8 pad)
    __shared__ __align__(16) bf16 Bs[2][128 * 40];

    const int tid  = threadIdx.x;
    const int wave = tid >> 5, lane = tid & 31;
    const int m    = lane & 15, half = lane >> 4;
    const int wm   = wave & 3;          // M direction, 32 rows each
    const int wn   = wave >> 2;         // N direction, 64 cols each
    const int row_base = blockIdx.y * 128;
    const int col_base = blockIdx.x * 128;

    v8f acc[2][4];
    #pragma unroll
    for (int mi = 0; mi < 2; ++mi)
        #pragma unroll
        for (int nj = 0; nj < 4; ++nj) acc[mi][nj] = vzero();

    // 512 16-byte chunks per matrix tile; 2 per thread per matrix
    auto issue_tile = [&](int buf, int kc) {
        #pragma unroll
        for (int p = 0; p < 2; ++p) {
            int ch  = tid + p * 256;
            int row = ch >> 2, q = ch & 3;
            cp16(&As[buf][row * 40 + q * 8],
                 &X[(size_t)(row_base + row) * K + kc + q * 8]);
            cp16(&Bs[buf][row * 40 + q * 8],
                 &W[(size_t)(col_base + row) * K + kc + q * 8]);
        }
    };

    issue_tile(0, 0);
    const int nk = K >> 5;
    for (int it = 0; it < nk; ++it) {
        const int cur = it & 1;
        async_wait();
        __syncthreads();
        if (it + 1 < nk) issue_tile(cur ^ 1, (it + 1) << 5);

        bf16x16 afrag[2], bfrag[4];
        #pragma unroll
        for (int mi = 0; mi < 2; ++mi) {
            const bf16* ap = &As[cur][(wm * 32 + mi * 16 + m) * 40];
            afrag[mi] = frag_from(ap + half * 8, ap + 16 + half * 8);
        }
        #pragma unroll
        for (int nj = 0; nj < 4; ++nj) {
            const bf16* bp = &Bs[cur][(wn * 64 + nj * 16 + m) * 40];
            bfrag[nj] = frag_from(bp + half * 8, bp + 16 + half * 8);
        }
        #pragma unroll
        for (int mi = 0; mi < 2; ++mi)
            #pragma unroll
            for (int nj = 0; nj < 4; ++nj)
                acc[mi][nj] = wmma_bf16(afrag[mi], bfrag[nj], acc[mi][nj]);
    }

    // epilogue: vgpr r holds row (r + 8*half), lane%16 holds col
    #pragma unroll
    for (int mi = 0; mi < 2; ++mi) {
        #pragma unroll
        for (int nj = 0; nj < 4; ++nj) {
            int col = col_base + wn * 64 + nj * 16 + m;
            float bv = bias[col];
            #pragma unroll
            for (int r = 0; r < 8; ++r) {
                int row = row_base + wm * 32 + mi * 16 + r + 8 * half;
                float v = acc[mi][nj][r] + bv;
                if (ACT == 1) v = gelu_f(v);
                if (RESID)    v += resid[(size_t)row * N + col];
                if (OUTBF) ((bf16*)Yv)[(size_t)row * N + col] = f2bf(v);
                else       ((float*)Yv)[(size_t)row * N + col] = v;
            }
        }
    }
}

// ---------------- flash attention (non-causal), bf16 in/out -----------------
// Grid: (T/64, B*H). Block: 128 threads = 4 waves; wave handles 16 query rows.
__global__ __launch_bounds__(128) void attn_kernel(const bf16* __restrict__ qkv,
                                                   bf16* __restrict__ o) {
    __shared__ __align__(16) bf16 Ks[64 * 72];        // [key][dim]
    __shared__ __align__(16) bf16 Vt[64 * 72];        // [dim][key]
    __shared__ __align__(16) bf16 Ps[4 * 16 * 72];    // per-wave P staging

    const int b    = blockIdx.y >> 4;        // H = 16
    const int h    = blockIdx.y & 15;
    const int q0   = blockIdx.x * 64;
    const int tid  = threadIdx.x;
    const int wave = tid >> 5, lane = tid & 31;
    const int m    = lane & 15, half = lane >> 4;
    const size_t C3 = (size_t)3 * C_;

    // Q fragments straight from global bf16 (scale folded into scores)
    bf16x16 aq[2];
    {
        const bf16* qp = qkv + (size_t)(b * T_ + q0 + wave * 16 + m) * C3 + h * HD_;
        #pragma unroll
        for (int ks = 0; ks < 2; ++ks)
            aq[ks] = frag_from(qp + ks * 32 + half * 8, qp + ks * 32 + 16 + half * 8);
    }

    v8f oacc[4];
    #pragma unroll
    for (int nj = 0; nj < 4; ++nj) oacc[nj] = vzero();
    float mi_[8], li_[8];
    #pragma unroll
    for (int r = 0; r < 8; ++r) { mi_[r] = -3.0e38f; li_[r] = 0.0f; }

    const bf16* kbase = qkv + (size_t)(b * T_) * C3 + C_ + h * HD_;

    for (int kb = 0; kb < T_; kb += 64) {
        __syncthreads();
        // K tile: 512 16B chunks via async copy (4 per thread)
        #pragma unroll
        for (int p = 0; p < 4; ++p) {
            int ch   = tid + p * 128;
            int krow = ch >> 3, q = ch & 7;
            cp16(&Ks[krow * 72 + q * 8], kbase + (size_t)(kb + krow) * C3 + q * 8);
        }
        // V tile transposed into LDS (bf16, register path)
        {
            int key = tid >> 1;
            int cg  = (tid & 1) * 32;
            const bf16* vp = qkv + (size_t)(b * T_ + kb + key) * C3 + 2 * C_ + h * HD_ + cg;
            #pragma unroll
            for (int j = 0; j < 4; ++j) {
                union { bf16x8 v; bf16 e[8]; } u;
                u.v = *(const bf16x8*)(vp + j * 8);
                #pragma unroll
                for (int e2 = 0; e2 < 8; ++e2)
                    Vt[(cg + j * 8 + e2) * 72 + key] = u.e[e2];
            }
        }
        async_wait();
        __syncthreads();

        // S = Q @ K^T  (16 x 64 per wave)
        v8f s[4];
        #pragma unroll
        for (int nj = 0; nj < 4; ++nj) s[nj] = vzero();
        #pragma unroll
        for (int nj = 0; nj < 4; ++nj) {
            const bf16* kpr = &Ks[(nj * 16 + m) * 72];
            #pragma unroll
            for (int ks = 0; ks < 2; ++ks) {
                bf16x16 bk = frag_from(kpr + ks * 32 + half * 8,
                                       kpr + ks * 32 + 16 + half * 8);
                s[nj] = wmma_bf16(aq[ks], bk, s[nj]);
            }
        }
        #pragma unroll
        for (int nj = 0; nj < 4; ++nj)
            #pragma unroll
            for (int r = 0; r < 8; ++r) s[nj][r] *= 0.125f;   // 1/sqrt(HD)

        // online softmax per row (row = r + 8*half)
        #pragma unroll
        for (int r = 0; r < 8; ++r) {
            float mx = fmaxf(fmaxf(s[0][r], s[1][r]), fmaxf(s[2][r], s[3][r]));
            #pragma unroll
            for (int off = 8; off >= 1; off >>= 1)
                mx = fmaxf(mx, __shfl_xor(mx, off, 32));
            float mnew  = fmaxf(mi_[r], mx);
            float alpha = __expf(mi_[r] - mnew);
            float rs = 0.0f;
            #pragma unroll
            for (int nj = 0; nj < 4; ++nj) {
                float p = __expf(s[nj][r] - mnew);
                s[nj][r] = p; rs += p;
            }
            #pragma unroll
            for (int off = 8; off >= 1; off >>= 1)
                rs += __shfl_xor(rs, off, 32);
            li_[r] = li_[r] * alpha + rs;
            mi_[r] = mnew;
            #pragma unroll
            for (int nj = 0; nj < 4; ++nj) oacc[nj][r] *= alpha;
            #pragma unroll
            for (int nj = 0; nj < 4; ++nj)
                Ps[(wave * 16 + r + 8 * half) * 72 + nj * 16 + m] = f2bf(s[nj][r]);
        }
        __syncthreads();

        // O += P @ V
        const bf16* ppr = &Ps[(wave * 16 + m) * 72];
        #pragma unroll
        for (int ks = 0; ks < 2; ++ks) {
            bf16x16 ap = frag_from(ppr + ks * 32 + half * 8,
                                   ppr + ks * 32 + 16 + half * 8);
            #pragma unroll
            for (int nj = 0; nj < 4; ++nj) {
                const bf16* vpr = &Vt[(nj * 16 + m) * 72];
                bf16x16 bv = frag_from(vpr + ks * 32 + half * 8,
                                       vpr + ks * 32 + 16 + half * 8);
                oacc[nj] = wmma_bf16(ap, bv, oacc[nj]);
            }
        }
    }

    // normalize and write bf16
    #pragma unroll
    for (int r = 0; r < 8; ++r) {
        float inv = 1.0f / li_[r];
        int qr = q0 + wave * 16 + r + 8 * half;
        #pragma unroll
        for (int nj = 0; nj < 4; ++nj)
            o[(size_t)(b * T_ + qr) * C_ + h * HD_ + nj * 16 + m] =
                f2bf(oacc[nj][r] * inv);
    }
}

// ---------------- lm_head: logits = xlnf @ wte^T (2 rows, fp32) -------------
__global__ __launch_bounds__(256) void lmhead_kernel(const float* __restrict__ xl,
                                                     const float* __restrict__ wte,
                                                     float* __restrict__ out) {
    int wave = threadIdx.x >> 5, lane = threadIdx.x & 31;
    int v = blockIdx.x * 8 + wave;
    if (v >= V_) return;
    const float* wr = wte + (size_t)v * C_;
    __builtin_prefetch(wr, 0, 0);                 // global_prefetch_b8
    float s0 = 0.0f, s1 = 0.0f;
    for (int k = lane; k < C_; k += 32) {
        float w = wr[k];
        s0 += w * xl[k];
        s1 += w * xl[C_ + k];
    }
    #pragma unroll
    for (int off = 16; off >= 1; off >>= 1) {
        s0 += __shfl_xor(s0, off, 32);
        s1 += __shfl_xor(s1, off, 32);
    }
    if (lane == 0) { out[v] = s0; out[V_ + v] = s1; }
}

// ---------------- driver ----------------
extern "C" void kernel_launch(void* const* d_in, const int* in_sizes, int n_in,
                              void* d_out, int out_size, void* d_ws, size_t ws_size,
                              hipStream_t stream) {
    (void)in_sizes; (void)n_in; (void)out_size; (void)ws_size;
    const int*   idx    = (const int*)  d_in[0];
    const float* wte    = (const float*)d_in[1];
    const float* wpe    = (const float*)d_in[2];
    const float* ln1_w  = (const float*)d_in[3];
    const float* ln1_b  = (const float*)d_in[4];
    const float* attn_w = (const float*)d_in[5];
    const float* attn_b = (const float*)d_in[6];
    const float* proj_w = (const float*)d_in[7];
    const float* proj_b = (const float*)d_in[8];
    const float* ln2_w  = (const float*)d_in[9];
    const float* ln2_b  = (const float*)d_in[10];
    const float* fc_w   = (const float*)d_in[11];
    const float* fc_b   = (const float*)d_in[12];
    const float* fc2_w  = (const float*)d_in[13];
    const float* fc2_b  = (const float*)d_in[14];
    const float* lnf_w  = (const float*)d_in[15];
    const float* lnf_b  = (const float*)d_in[16];

    const size_t MT = (size_t)B_ * T_;          // 2048 rows

    // workspace carve: fp32 region, then bf16 region
    float* x    = (float*)d_ws;                 // [MT, C] residual (fp32)
    float* xlnf = x + MT * C_;                  // [2, C]  (fp32)
    bf16*  bb   = (bf16*)(xlnf + 2 * C_);
    bf16* ha    = bb; bb += MT * C_;            // LN outputs
    bf16* qkvb  = bb; bb += MT * 3 * C_;        // QKV (bf16)
    bf16* ob    = bb; bb += MT * C_;            // attention out
    bf16* h2b   = bb; bb += MT * 4 * C_;        // GELU out
    bf16* attn_wb = bb; bb += (size_t)L_ * 3 * C_ * C_;
    bf16* proj_wb = bb; bb += (size_t)L_ * C_ * C_;
    bf16* fc_wb   = bb; bb += (size_t)L_ * 4 * C_ * C_;
    bf16* fc2_wb  = bb; bb += (size_t)L_ * 4 * C_ * C_;

    // pre-convert weights fp32 -> bf16 (one pass each)
    {
        int n4;
        n4 = L_ * 3 * C_ * C_ / 4;
        wconv_kernel<<<(n4 + 255) / 256, 256, 0, stream>>>(attn_w, attn_wb, n4);
        n4 = L_ * C_ * C_ / 4;
        wconv_kernel<<<(n4 + 255) / 256, 256, 0, stream>>>(proj_w, proj_wb, n4);
        n4 = L_ * 4 * C_ * C_ / 4;
        wconv_kernel<<<(n4 + 255) / 256, 256, 0, stream>>>(fc_w, fc_wb, n4);
        n4 = L_ * 4 * C_ * C_ / 4;
        wconv_kernel<<<(n4 + 255) / 256, 256, 0, stream>>>(fc2_w, fc2_wb, n4);
    }

    embed_kernel<<<(int)MT, 256, 0, stream>>>(idx, wte, wpe, x);

    for (int l = 0; l < L_; ++l) {
        ln_kernel<true><<<(int)MT, 256, 0, stream>>>(x, ha, ln1_w + l * C_,
                                                     ln1_b + l * C_, (size_t)C_);
        gemm_kernel<0, false, true><<<dim3(3 * C_ / 128, (int)MT / 128), 256, 0, stream>>>(
            ha, attn_wb + (size_t)l * 3 * C_ * C_, attn_b + l * 3 * C_,
            nullptr, qkvb, (int)MT, 3 * C_, C_);
        attn_kernel<<<dim3(T_ / 64, B_ * H_), 128, 0, stream>>>(qkvb, ob);
        gemm_kernel<0, true, false><<<dim3(C_ / 128, (int)MT / 128), 256, 0, stream>>>(
            ob, proj_wb + (size_t)l * C_ * C_, proj_b + l * C_,
            x, x, (int)MT, C_, C_);
        ln_kernel<true><<<(int)MT, 256, 0, stream>>>(x, ha, ln2_w + l * C_,
                                                     ln2_b + l * C_, (size_t)C_);
        gemm_kernel<1, false, true><<<dim3(4 * C_ / 128, (int)MT / 128), 256, 0, stream>>>(
            ha, fc_wb + (size_t)l * 4 * C_ * C_, fc_b + l * 4 * C_,
            nullptr, h2b, (int)MT, 4 * C_, C_);
        gemm_kernel<0, true, false><<<dim3(C_ / 128, (int)MT / 128), 256, 0, stream>>>(
            h2b, fc2_wb + (size_t)l * 4 * C_ * C_, fc2_b + l * C_,
            x, x, (int)MT, C_, 4 * C_);
    }

    // final LN on the two last-position rows only (rows T-1 and 2T-1)
    ln_kernel<false><<<2, 256, 0, stream>>>(x + (size_t)(T_ - 1) * C_, xlnf,
                                            lnf_w, lnf_b, (size_t)T_ * C_);
    lmhead_kernel<<<(V_ + 7) / 8, 256, 0, stream>>>(xlnf, wte, (float*)d_out);
}
